// CausalSelfAttention_15504831939141
// MI455X (gfx1250) — matrix operand order repaired and checked by reading, not compiled, not run on hardware
//
#include <hip/hip_runtime.h>

// Problem constants (from reference): B=2, T=4096, C=768, H=12, D=64
#define BB 2
#define TT 4096
#define CC 768
#define HH 12
#define DD 64

typedef __attribute__((ext_vector_type(16))) __bf16 v16bf;
typedef __attribute__((ext_vector_type(8)))  float  v8f;

struct alignas(16) U128 { unsigned x, y, z, w; };
struct alignas(16) F128 { float x, y, z, w; };

union AFrag {
    v16bf v;
    unsigned short s[16];
    U128 q[2];
};

union H8 {
    U128 q;
    unsigned short s[8];
};

// ---- optional CDNA5 async global->LDS path (guarded; falls back cleanly) ----
#if __has_builtin(__builtin_amdgcn_global_load_async_to_lds_b128) && \
    __has_builtin(__builtin_amdgcn_s_wait_asynccnt)
#define HAVE_ASYNC_LDS 1
#else
#define HAVE_ASYNC_LDS 0
#endif

#if HAVE_ASYNC_LDS
typedef int i32x4 __attribute__((vector_size(16)));

__device__ __forceinline__ void async_copy_b128(const unsigned short* g, unsigned short* l) {
    __builtin_amdgcn_global_load_async_to_lds_b128(
        (__attribute__((address_space(1))) i32x4*)g,
        (__attribute__((address_space(3))) i32x4*)l,
        0, 0);
}
__device__ __forceinline__ void async_wait0() {
    __builtin_amdgcn_s_wait_asynccnt(0);
}
#endif

__device__ __forceinline__ unsigned short f2bf(float f) {
    union { float f; unsigned u; } x; x.f = f;
    unsigned r = x.u + 0x7FFFu + ((x.u >> 16) & 1u);
    return (unsigned short)(r >> 16);
}

// A-matrix fragment (16x32 bf16, row-major tile in LDS).
// Lane holds M = lane&15; elements e -> K = half*16 + e (contiguous)
__device__ __forceinline__ v16bf load_afrag(const unsigned short* tile, int stride, int lane) {
    const int row  = lane & 15;
    const int half = lane >> 4;
    AFrag u;
    const unsigned short* p = tile + row * stride;
    u.q[0] = *(const U128*)(p + (half << 4));
    u.q[1] = *(const U128*)(p + 8 + (half << 4));
    return u.v;
}

// B-matrix fragment (32x16 bf16 K x N), staged N-major in LDS: tileT[n*stride + k].
// Lane holds N = lane&15; elements e -> K = (lane>>4)*16 + e (contiguous).
__device__ __forceinline__ v16bf load_bfrag(const unsigned short* tileT, int stride, int lane) {
    const int n    = lane & 15;
    const int half = lane >> 4;
    AFrag u;
    const unsigned short* p = tileT + n * stride + (half << 4);
    u.q[0] = *(const U128*)(p);
    u.q[1] = *(const U128*)(p + 8);
    return u.v;
}

// ---------------------------------------------------------------------------
// fp32 -> bf16 elementwise conversion (n must be a multiple of 4)
// ---------------------------------------------------------------------------
__global__ __launch_bounds__(256) void f32_to_bf16_kernel(const float* __restrict__ src,
                                                          unsigned short* __restrict__ dst,
                                                          int n) {
    int i = (blockIdx.x * 256 + threadIdx.x) * 4;
    if (i + 3 < n) {
        F128 f = *(const F128*)(src + i);
        union { unsigned short s[4]; unsigned long long u; } o;
        o.s[0] = f2bf(f.x); o.s[1] = f2bf(f.y); o.s[2] = f2bf(f.z); o.s[3] = f2bf(f.w);
        *(unsigned long long*)(dst + i) = o.u;
    }
}

// ---------------------------------------------------------------------------
// Generic bf16 WMMA GEMM: C[M,N] = A[M,K] * B[K,N] + bias[N]
// mode 0: scatter bf16 output into Q/K/V (B,H,T,D) buffers (N = 3C)
// mode 1: write fp32 output row-major (final projection)
// Block tile 128x128, BK=32, 256 threads = 8 waves (4 along M x 2 along N),
// each wave computes a 32x64 sub-tile = 2x4 WMMA accumulators.
// ---------------------------------------------------------------------------
__global__ __launch_bounds__(256) void gemm_bf16_kernel(
    const unsigned short* __restrict__ A,
    const unsigned short* __restrict__ Bm,
    const float* __restrict__ bias,
    float* __restrict__ outF,
    unsigned short* __restrict__ outQ,
    unsigned short* __restrict__ outK,
    unsigned short* __restrict__ outV,
    int M, int N, int K, int mode)
{
    __shared__ __align__(16) unsigned short As [128 * 40];  // row-major, stride 40
    __shared__ __align__(16) unsigned short BsT[128 * 40];  // N-major (transposed), stride 40

    const int tid  = threadIdx.x;
    const int lane = tid & 31;
    const int wid  = tid >> 5;
    const int wm   = wid & 3;   // wave m-offset = wm*32
    const int wn   = wid >> 2;  // wave n-offset = wn*64
    const int bm   = blockIdx.y * 128;
    const int bn   = blockIdx.x * 128;

    v8f acc[2][4];
    const v8f vzero = {0.f, 0.f, 0.f, 0.f, 0.f, 0.f, 0.f, 0.f};
#pragma unroll
    for (int i = 0; i < 2; ++i)
#pragma unroll
        for (int j = 0; j < 4; ++j) acc[i][j] = vzero;

    for (int k0 = 0; k0 < K; k0 += 32) {
        // ---- stage A tile: 128 rows x 32 k (verbatim copy -> async path) ----
        {
            const int row  = tid >> 1;
            const int koff = (tid & 1) << 4;
            const unsigned short* g = A + (size_t)(bm + row) * K + k0 + koff;
            if (k0 + 32 < K) __builtin_prefetch(g + 32, 0, 1);
#if HAVE_ASYNC_LDS
            async_copy_b128(g,     &As[row * 40 + koff]);
            async_copy_b128(g + 8, &As[row * 40 + koff + 8]);
#else
            U128 d0 = *(const U128*)g;
            U128 d1 = *(const U128*)(g + 8);
            *(U128*)&As[row * 40 + koff]     = d0;
            *(U128*)&As[row * 40 + koff + 8] = d1;
#endif
        }
        // ---- stage B tile transposed: global (K x N) -> LDS (N x K) ----
        {
            const int kk  = tid & 31;
            const int nch = tid >> 5;  // 8 chunks of 16 columns
            const unsigned short* g = Bm + (size_t)(k0 + kk) * N + bn + nch * 16;
            if (k0 + 32 < K) __builtin_prefetch(g + (size_t)32 * N, 0, 1);
            H8 u0, u1;
            u0.q = *(const U128*)g;
            u1.q = *(const U128*)(g + 8);
#pragma unroll
            for (int i = 0; i < 8; ++i) {
                BsT[(nch * 16 + i) * 40 + kk]     = u0.s[i];
                BsT[(nch * 16 + 8 + i) * 40 + kk] = u1.s[i];
            }
        }
#if HAVE_ASYNC_LDS
        async_wait0();
#endif
        __syncthreads();

        const v16bf a0 = load_afrag(&As[(wm * 32) * 40], 40, lane);
        const v16bf a1 = load_afrag(&As[(wm * 32 + 16) * 40], 40, lane);
#pragma unroll
        for (int c = 0; c < 4; ++c) {
            const v16bf bf = load_bfrag(&BsT[(wn * 64 + c * 16) * 40], 40, lane);
            acc[0][c] = __builtin_amdgcn_wmma_f32_16x16x32_bf16(
                false, a0, false, bf, (short)0, acc[0][c], false, false);
            acc[1][c] = __builtin_amdgcn_wmma_f32_16x16x32_bf16(
                false, a1, false, bf, (short)0, acc[1][c], false, false);
        }
        __syncthreads();
    }

    // ---- epilogue ----
    const int half = lane >> 4;
    const int ln   = lane & 15;
#pragma unroll
    for (int tr = 0; tr < 2; ++tr) {
#pragma unroll
        for (int c = 0; c < 4; ++c) {
#pragma unroll
            for (int r = 0; r < 8; ++r) {
                const int m = bm + wm * 32 + tr * 16 + r + half * 8;
                const int n = bn + wn * 64 + c * 16 + ln;
                const float v = acc[tr][c][r] + bias[n];
                if (mode == 1) {
                    outF[(size_t)m * N + n] = v;
                } else {
                    const int sel = n / CC;
                    const int nc  = n - sel * CC;
                    const int h   = nc >> 6;
                    const int d   = nc & 63;
                    const int bi  = m >> 12;         // / T
                    const int t   = m & (TT - 1);    // % T
                    const size_t idx = (((size_t)bi * HH + h) * TT + t) * DD + d;
                    const unsigned short bv = f2bf(v);
                    if (sel == 0)      outQ[idx] = bv;
                    else if (sel == 1) outK[idx] = bv;
                    else               outV[idx] = bv;
                }
            }
        }
    }
}

// ---------------------------------------------------------------------------
// Flash attention (causal): one block = 64 query rows of one (b,h); 4 waves,
// each wave owns a 16-row query tile. K tiles (32 keys x 64d) and transposed
// V tiles staged in LDS. Online softmax with 16-lane shuffle reductions.
// ---------------------------------------------------------------------------
__global__ __launch_bounds__(128) void flash_attn_kernel(
    const unsigned short* __restrict__ Qm,
    const unsigned short* __restrict__ Km,
    const unsigned short* __restrict__ Vm,
    unsigned short* __restrict__ Y)
{
    __shared__ __align__(16) unsigned short Ks[32 * 72];      // key-major, stride 72
    __shared__ __align__(16) unsigned short VT[64 * 40];      // d-major (transposed), stride 40
    __shared__ __align__(16) unsigned short Ps[4 * 16 * 32];  // per-wave P staging, stride 32

    const int tid  = threadIdx.x;
    const int lane = tid & 31;
    const int wid  = tid >> 5;
    const int ln   = lane & 15;
    const int half = lane >> 4;

    const int ntq = TT / 64;
    const int qt  = blockIdx.x % ntq;
    const int bh  = blockIdx.x / ntq;

    const size_t base  = (size_t)bh * TT * DD;
    const int    qbase = qt * 64 + wid * 16;

    // Q fragments: A-layout, contraction d split 0..31 / 32..63
    const unsigned short* qp = Qm + base + (size_t)(qbase + ln) * DD;
    AFrag q0u, q1u;
    q0u.q[0] = *(const U128*)(qp + half * 16);
    q0u.q[1] = *(const U128*)(qp + 8 + half * 16);
    q1u.q[0] = *(const U128*)(qp + 32 + half * 16);
    q1u.q[1] = *(const U128*)(qp + 40 + half * 16);

    const v8f vzero = {0.f, 0.f, 0.f, 0.f, 0.f, 0.f, 0.f, 0.f};
    v8f o[4];
    float mrow[8], lrow[8];
#pragma unroll
    for (int c = 0; c < 4; ++c) o[c] = vzero;
#pragma unroll
    for (int r = 0; r < 8; ++r) { mrow[r] = -1e30f; lrow[r] = 0.f; }

    const float scale = 0.125f;  // 1/sqrt(64)
    const int nkt = 2 * qt + 2;  // key tiles of 32 covering causal extent

    for (int j = 0; j < nkt; ++j) {
        const int kbase = j * 32;
        // ---- stage K tile (verbatim -> async path) and V tile (transposed) ----
        {
            const int key  = tid >> 2;
            const int doff = (tid & 3) << 4;
            const unsigned short* gk = Km + base + (size_t)(kbase + key) * DD + doff;
#if HAVE_ASYNC_LDS
            async_copy_b128(gk,     &Ks[key * 72 + doff]);
            async_copy_b128(gk + 8, &Ks[key * 72 + doff + 8]);
#else
            U128 d0 = *(const U128*)gk;
            U128 d1 = *(const U128*)(gk + 8);
            *(U128*)&Ks[key * 72 + doff]     = d0;
            *(U128*)&Ks[key * 72 + doff + 8] = d1;
#endif

            const unsigned short* gv = Vm + base + (size_t)(kbase + key) * DD + doff;
            H8 u0, u1;
            u0.q = *(const U128*)gv;
            u1.q = *(const U128*)(gv + 8);
#pragma unroll
            for (int i = 0; i < 8; ++i) {
                VT[(doff + i) * 40 + key]     = u0.s[i];
                VT[(doff + 8 + i) * 40 + key] = u1.s[i];
            }
        }
#if HAVE_ASYNC_LDS
        async_wait0();
#endif
        __syncthreads();

        if (kbase <= qbase + 15) {
            // ---- S = Q * K^T (two key subtiles, contraction over d in 2 WMMAs) ----
            v8f s0 = vzero, s1 = vzero;
#pragma unroll
            for (int ds = 0; ds < 2; ++ds) {
                AFrag b0, b1;
                const unsigned short* p0 = &Ks[ln * 72 + ds * 32 + half * 16];
                b0.q[0] = *(const U128*)p0;
                b0.q[1] = *(const U128*)(p0 + 8);
                const unsigned short* p1 = &Ks[(16 + ln) * 72 + ds * 32 + half * 16];
                b1.q[0] = *(const U128*)p1;
                b1.q[1] = *(const U128*)(p1 + 8);
                const v16bf aa = ds ? q1u.v : q0u.v;
                s0 = __builtin_amdgcn_wmma_f32_16x16x32_bf16(false, aa, false, b0.v,
                                                             (short)0, s0, false, false);
                s1 = __builtin_amdgcn_wmma_f32_16x16x32_bf16(false, aa, false, b1.v,
                                                             (short)0, s1, false, false);
            }

            // ---- online softmax: a row spans the 16 lanes of this half-wave ----
#pragma unroll
            for (int r = 0; r < 8; ++r) {
                const int row = qbase + r + half * 8;
                float v0 = (kbase + ln)      <= row ? s0[r] * scale : -1e30f;
                float v1 = (kbase + 16 + ln) <= row ? s1[r] * scale : -1e30f;
                float tmax = fmaxf(v0, v1);
#pragma unroll
                for (int off = 1; off < 16; off <<= 1)
                    tmax = fmaxf(tmax, __shfl_xor(tmax, off, 32));
                const float mn    = fmaxf(mrow[r], tmax);
                const float alpha = __expf(mrow[r] - mn);
                const float p0    = __expf(v0 - mn);
                const float p1    = __expf(v1 - mn);
                float rs = p0 + p1;
#pragma unroll
                for (int off = 1; off < 16; off <<= 1)
                    rs += __shfl_xor(rs, off, 32);
                mrow[r] = mn;
                lrow[r] = lrow[r] * alpha + rs;
#pragma unroll
                for (int c = 0; c < 4; ++c) o[c][r] *= alpha;
                // stage P (bf16) into this wave's LDS region, row-major 16x32
                unsigned short* pr = &Ps[(wid * 16 + r + half * 8) * 32];
                pr[ln]      = f2bf(p0);
                pr[16 + ln] = f2bf(p1);
            }

            // ---- O += P * V ----
            const v16bf pf = load_afrag(&Ps[wid * 16 * 32], 32, lane);
#pragma unroll
            for (int c = 0; c < 4; ++c) {
                AFrag vb;
                const unsigned short* pv = &VT[(c * 16 + ln) * 40 + half * 16];
                vb.q[0] = *(const U128*)pv;
                vb.q[1] = *(const U128*)(pv + 8);
                o[c] = __builtin_amdgcn_wmma_f32_16x16x32_bf16(false, pf, false, vb.v,
                                                               (short)0, o[c], false, false);
            }
        }
        __syncthreads();
    }

    // ---- epilogue: Y (B,T,C) bf16, c = h*64 + d ----
    const int b = bh / HH, h = bh % HH;
#pragma unroll
    for (int r = 0; r < 8; ++r) {
        const int   t    = qbase + r + half * 8;
        const float inv  = 1.0f / lrow[r];
        const size_t rb  = ((size_t)b * TT + t) * CC + h * 64;
#pragma unroll
        for (int c = 0; c < 4; ++c)
            Y[rb + c * 16 + ln] = f2bf(o[c][r] * inv);
    }
}

// ---------------------------------------------------------------------------
extern "C" void kernel_launch(void* const* d_in, const int* in_sizes, int n_in,
                              void* d_out, int out_size, void* d_ws, size_t ws_size,
                              hipStream_t stream) {
    (void)in_sizes; (void)n_in; (void)out_size; (void)ws_size;

    const float* x      = (const float*)d_in[0];
    const float* w_attn = (const float*)d_in[1];
    const float* b_attn = (const float*)d_in[2];
    const float* w_proj = (const float*)d_in[3];
    const float* b_proj = (const float*)d_in[4];
    float* out = (float*)d_out;

    const size_t nX  = (size_t)BB * TT * CC;      // 6,291,456
    const size_t nWA = (size_t)CC * 3 * CC;       // 1,769,472
    const size_t nWP = (size_t)CC * CC;           //   589,824
    const size_t nHD = (size_t)BB * HH * TT * DD; // 6,291,456

    char* w = (char*)d_ws;
    size_t off = 0;
    auto alloc = [&](size_t bytes) -> char* {
        char* p = w + off;
        off += (bytes + 255) & ~(size_t)255;
        return p;
    };
    unsigned short* xb  = (unsigned short*)alloc(nX * 2);
    unsigned short* wab = (unsigned short*)alloc(nWA * 2);
    unsigned short* wpb = (unsigned short*)alloc(nWP * 2);
    unsigned short* qb  = (unsigned short*)alloc(nHD * 2);
    unsigned short* kb  = (unsigned short*)alloc(nHD * 2);
    unsigned short* vb  = (unsigned short*)alloc(nHD * 2);
    unsigned short* yb  = (unsigned short*)alloc(nX * 2);

    // 1) fp32 -> bf16 conversions
    f32_to_bf16_kernel<<<(int)(nX / 4 / 256), 256, 0, stream>>>(x, xb, (int)nX);
    f32_to_bf16_kernel<<<(int)(nWA / 4 / 256), 256, 0, stream>>>(w_attn, wab, (int)nWA);
    f32_to_bf16_kernel<<<(int)(nWP / 4 / 256), 256, 0, stream>>>(w_proj, wpb, (int)nWP);

    // 2) QKV projection GEMM: (B*T, C) x (C, 3C) -> q/k/v bf16 (B,H,T,D)
    {
        dim3 grid(3 * CC / 128, BB * TT / 128);
        gemm_bf16_kernel<<<grid, 256, 0, stream>>>(
            xb, wab, b_attn, nullptr, qb, kb, vb,
            BB * TT, 3 * CC, CC, 0);
    }

    // 3) causal flash attention -> yb bf16 (B,T,C)
    {
        const int nblk = BB * HH * (TT / 64);
        flash_attn_kernel<<<nblk, 128, 0, stream>>>(qb, kb, vb, yb);
    }

    // 4) output projection GEMM: (B*T, C) x (C, C) + b_proj -> fp32 out
    {
        dim3 grid(CC / 128, BB * TT / 128);
        gemm_bf16_kernel<<<grid, 256, 0, stream>>>(
            yb, wpb, b_proj, out, nullptr, nullptr, nullptr,
            BB * TT, CC, CC, 1);
    }
}
